// DRMAttention_21431886807254
// MI455X (gfx1250) — matrix operand order repaired and checked by reading, not compiled, not run on hardware
//
#include <hip/hip_runtime.h>

// ---------------- Problem constants ----------------
#define Bc 2
#define Tt 512
#define DMc 1024
#define Hc 16
#define DHc 64
#define Dc 32
#define Rc 4

typedef __attribute__((ext_vector_type(16))) _Float16 v16h;
typedef __attribute__((ext_vector_type(4)))  _Float16 v4h;
typedef __attribute__((ext_vector_type(8)))  float    v8f;
typedef __attribute__((ext_vector_type(4)))  float    v4f;

typedef unsigned int u32x4 __attribute__((ext_vector_type(4)));
typedef int          i32x4 __attribute__((ext_vector_type(4)));
typedef int          i32x8 __attribute__((ext_vector_type(8)));

#if defined(__has_builtin)
#  if __has_builtin(__builtin_amdgcn_tensor_load_to_lds)
#    define HAVE_TDM 1
#  else
#    define HAVE_TDM 0
#  endif
#else
#  define HAVE_TDM 0
#endif

union HF16 { v16h v; _Float16 h[16]; };

// ---- WMMA fragment helpers (layouts per CDNA5 ISA 7.12.2) ----
__device__ __forceinline__ v16h load_fragA(const _Float16* base, int ld, int lane) {
    const int m  = lane & 15;
    const int kb = (lane < 16) ? 0 : 8;
    const _Float16* row = base + (size_t)m * ld + kb;
    HF16 f;
#pragma unroll
    for (int e = 0; e < 8; ++e) f.h[e] = row[e];
#pragma unroll
    for (int e = 0; e < 8; ++e) f.h[8 + e] = row[16 + e];
    return f.v;
}

__device__ __forceinline__ v16h load_fragB(const _Float16* baseT, int ld, int lane) {
    const int n  = lane & 15;
    const int kb = (lane < 16) ? 0 : 8;
    const _Float16* row = baseT + (size_t)n * ld + kb;
    HF16 f;
#pragma unroll
    for (int e = 0; e < 8; ++e) f.h[e] = row[e];
#pragma unroll
    for (int e = 0; e < 8; ++e) f.h[8 + e] = row[16 + e];
    return f.v;
}

__device__ __forceinline__ v8f wmma_f16(v16h a, v16h b, v8f c) {
    return __builtin_amdgcn_wmma_f32_16x16x32_f16(
        false, a, false, b, (short)0, c, false, false);
}

__device__ __forceinline__ float redmax16(float v) {
    v = fmaxf(v, __shfl_xor(v, 1, 32));
    v = fmaxf(v, __shfl_xor(v, 2, 32));
    v = fmaxf(v, __shfl_xor(v, 4, 32));
    v = fmaxf(v, __shfl_xor(v, 8, 32));
    return v;
}
__device__ __forceinline__ float redsum16(float v) {
    v += __shfl_xor(v, 1, 32);
    v += __shfl_xor(v, 2, 32);
    v += __shfl_xor(v, 4, 32);
    v += __shfl_xor(v, 8, 32);
    return v;
}

// ===================================================================
// GEMM: Y[M,N] = X[M,K] * W[K,N] via f16 WMMA. 256 thr (8 waves),
// 128x128 tile, k-step 32, wave = 64x32 => 8 WMMAs/step.
// Double-buffered LDS, software-pipelined: next-tile global loads are
// issued before the WMMAs, converted/stored after; 1 barrier per step.
// ===================================================================
__global__ __launch_bounds__(256)
void drm_gemm_f16w(const float* __restrict__ X, const float* __restrict__ W,
                   float* __restrict__ Y, int M, int N, int K)
{
    __shared__ _Float16 Xs[2][128][40];   // [buf][m][k]
    __shared__ _Float16 Wts[2][128][40];  // [buf][n][k] (transposed)

    const int tid  = threadIdx.x;
    const int lane = tid & 31;
    const int wid  = tid >> 5;
    const int ntiles = N / 128;
    const int m0 = (blockIdx.x / ntiles) * 128;
    const int n0 = (blockIdx.x % ntiles) * 128;
    const int wm = (wid >> 2) * 64;
    const int wn = (wid & 3) * 32;

    // fixed per-thread staging coordinates (4 float4 each for X and W)
    int xm[4], xk[4], wkk[4], wn4[4];
#pragma unroll
    for (int it = 0; it < 4; ++it) {
        int i4 = tid + it * 256;
        xm[it]  = i4 >> 3;          // 0..127
        xk[it]  = (i4 & 7) * 4;     // 0..28
        wkk[it] = i4 >> 5;          // 0..31
        wn4[it] = (i4 & 31) * 4;    // 0..124
    }

    v8f C[4][2];
#pragma unroll
    for (int a = 0; a < 4; ++a)
#pragma unroll
        for (int b = 0; b < 2; ++b)
#pragma unroll
            for (int e = 0; e < 8; ++e) C[a][b][e] = 0.0f;

    // ---- prologue: stage k-tile 0 into buffer 0 ----
    {
        v4f xr[4], wr[4];
#pragma unroll
        for (int it = 0; it < 4; ++it)
            xr[it] = *(const v4f*)(X + (size_t)(m0 + xm[it]) * K + xk[it]);
#pragma unroll
        for (int it = 0; it < 4; ++it)
            wr[it] = *(const v4f*)(W + (size_t)wkk[it] * N + n0 + wn4[it]);
#pragma unroll
        for (int it = 0; it < 4; ++it) {
            v4h h4;
#pragma unroll
            for (int e = 0; e < 4; ++e) h4[e] = (_Float16)xr[it][e];
            *(v4h*)&Xs[0][xm[it]][xk[it]] = h4;
        }
#pragma unroll
        for (int it = 0; it < 4; ++it)
#pragma unroll
            for (int e = 0; e < 4; ++e)
                Wts[0][wn4[it] + e][wkk[it]] = (_Float16)wr[it][e];
    }
    __syncthreads();

    for (int k0 = 0; k0 < K; k0 += 32) {
        const int  cur  = (k0 >> 5) & 1;
        const int  nxt  = cur ^ 1;
        const bool more = (k0 + 32) < K;

        // 1) issue next-tile global loads (latency hidden by WMMAs below)
        v4f xr[4], wr[4];
        if (more) {
            const int kn = k0 + 32;
#pragma unroll
            for (int it = 0; it < 4; ++it)
                xr[it] = *(const v4f*)(X + (size_t)(m0 + xm[it]) * K + kn + xk[it]);
#pragma unroll
            for (int it = 0; it < 4; ++it)
                wr[it] = *(const v4f*)(W + (size_t)(kn + wkk[it]) * N + n0 + wn4[it]);
            if (k0 + 64 < K) {
                __builtin_prefetch(X + (size_t)(m0 + xm[0]) * K + (k0 + 64) + xk[0], 0, 1);
                __builtin_prefetch(W + (size_t)(k0 + 64 + wkk[0]) * N + n0 + wn4[0], 0, 1);
            }
        }

        // 2) compute on current buffer
        v16h A[4], Bf[2];
#pragma unroll
        for (int a = 0; a < 4; ++a)
            A[a] = load_fragA(&Xs[cur][wm + a * 16][0], 40, lane);
#pragma unroll
        for (int b = 0; b < 2; ++b)
            Bf[b] = load_fragB(&Wts[cur][wn + b * 16][0], 40, lane);
#pragma unroll
        for (int a = 0; a < 4; ++a)
#pragma unroll
            for (int b = 0; b < 2; ++b)
                C[a][b] = wmma_f16(A[a], Bf[b], C[a][b]);

        // 3) convert + store next tile into the other buffer
        if (more) {
#pragma unroll
            for (int it = 0; it < 4; ++it) {
                v4h h4;
#pragma unroll
                for (int e = 0; e < 4; ++e) h4[e] = (_Float16)xr[it][e];
                *(v4h*)&Xs[nxt][xm[it]][xk[it]] = h4;
            }
#pragma unroll
            for (int it = 0; it < 4; ++it)
#pragma unroll
                for (int e = 0; e < 4; ++e)
                    Wts[nxt][wn4[it] + e][wkk[it]] = (_Float16)wr[it][e];
        }
        __syncthreads();
    }

    const int n16 = lane & 15, hi = lane >> 4;
#pragma unroll
    for (int a = 0; a < 4; ++a)
#pragma unroll
        for (int b = 0; b < 2; ++b)
#pragma unroll
            for (int e = 0; e < 8; ++e) {
                int row = m0 + wm + a * 16 + e + 8 * hi;
                int col = n0 + wn + b * 16 + n16;
                Y[(size_t)row * N + col] = C[a][b][e];
            }
}

// ===================================================================
// Kernel 2: per-token metric precompute (VALU). One thread per (b,h,t).
// ===================================================================
__global__ __launch_bounds__(256)
void drm_metric(const float* __restrict__ Qb, const float* __restrict__ Kb,
                const float* __restrict__ Wqm, const float* __restrict__ Wkm,
                const float* __restrict__ Wmet,
                _Float16* __restrict__ qa_h, _Float16* __restrict__ km_h,
                _Float16* __restrict__ U_h,
                float* __restrict__ rowbias, float* __restrict__ k2g)
{
    __shared__ float sWqm[DHc * Dc];
    __shared__ float sWkm[DHc * Dc];
    __shared__ float sWm[Dc * Dc * Rc];
    for (int i = threadIdx.x; i < DHc * Dc; i += 256) { sWqm[i] = Wqm[i]; sWkm[i] = Wkm[i]; }
    for (int i = threadIdx.x; i < Dc * Dc * Rc; i += 256) sWm[i] = Wmet[i];
    __syncthreads();

    const int g = blockIdx.x * 256 + threadIdx.x;
    const int t = g % Tt;
    const int h = (g / Tt) % Hc;
    const int b = g / (Tt * Hc);
    const float pos = (float)t;

    const float* qrow = Qb + (size_t)(b * Tt + t) * DMc + h * DHc;
    const float* krow = Kb + (size_t)(b * Tt + t) * DMc + h * DHc;

    float qm[Dc], km[Dc];
    {
        float qr[DHc];
#pragma unroll
        for (int j = 0; j < 32; ++j) {
            float f  = __expf(-(2.0f * j / 64.0f) * 9.210340371976184f);
            float cs = __cosf(pos * f), sn = __sinf(pos * f);
            float a = qrow[j], bq = qrow[j + 32];
            qr[j] = a * cs - bq * sn;
            qr[j + 32] = a * sn + bq * cs;
        }
        for (int d = 0; d < Dc; ++d) {
            float acc = 0.0f;
#pragma unroll
            for (int dh = 0; dh < DHc; ++dh) acc += qr[dh] * sWqm[dh * Dc + d];
            qm[d] = 1.0f / (1.0f + __expf(-acc));
        }
    }
    {
        float kr[DHc];
#pragma unroll
        for (int j = 0; j < 32; ++j) {
            float f  = __expf(-(2.0f * j / 64.0f) * 9.210340371976184f);
            float cs = __cosf(pos * f), sn = __sinf(pos * f);
            float a = krow[j], bk = krow[j + 32];
            kr[j] = a * cs - bk * sn;
            kr[j + 32] = a * sn + bk * cs;
        }
        for (int d = 0; d < Dc; ++d) {
            float acc = 0.0f;
#pragma unroll
            for (int dh = 0; dh < DHc; ++dh) acc += kr[dh] * sWkm[dh * Dc + d];
            km[d] = 1.0f / (1.0f + __expf(-acc));
        }
    }

    float q2 = 0.0f, k2 = 0.0f;
    for (int d = 0; d < Dc; ++d) { q2 += qm[d] * qm[d]; k2 += km[d] * km[d]; }

    float Uq[Rc] = {0.f, 0.f, 0.f, 0.f};
    for (int d = 0; d < Dc; ++d) {
#pragma unroll
        for (int r = 0; r < Rc; ++r) {
            float u = 0.0f;
            for (int m = 0; m < Dc; ++m) u += qm[m] * sWm[m * (Dc * Rc) + d * Rc + r];
            Uq[r] += u * qm[d];
        }
    }
    float sumUq2 = 0.0f;
#pragma unroll
    for (int r = 0; r < Rc; ++r) sumUq2 += Uq[r] * Uq[r];

    const int bh = b * Hc + h;
    const size_t base = ((size_t)bh * Tt + t) * Dc;
    for (int d = 0; d < Dc; ++d) {
        float cd = 0.0f;
#pragma unroll
        for (int r = 0; r < Rc; ++r) {
            float u = 0.0f;
            for (int m = 0; m < Dc; ++m) u += qm[m] * sWm[m * (Dc * Rc) + d * Rc + r];
            cd += Uq[r] * u;
            U_h[(((size_t)bh * Rc + r) * Tt + t) * Dc + d] = (_Float16)u;
        }
        qa_h[base + d] = (_Float16)(qm[d] + cd);
        km_h[base + d] = (_Float16)km[d];
    }
    rowbias[bh * Tt + t] = q2 + sumUq2;
    k2g[bh * Tt + t]     = k2;
}

// ===================================================================
// Kernel 3: flash-style metric attention (TDM-staged km tiles).
// ===================================================================
#define KMLD 36   // km LDS row stride in halves (64B data + 8B TDM pad)

__global__ __launch_bounds__(128)
void drm_attn(const _Float16* __restrict__ qa_h, const _Float16* __restrict__ km_h,
              const _Float16* __restrict__ U_h,
              const float* __restrict__ rowbias, const float* __restrict__ k2g,
              const float* __restrict__ Vbuf, const float* __restrict__ temp_p,
              float* __restrict__ attn_out)
{
    const int tiles = Tt / 64;
    const int bh = blockIdx.x / tiles;
    const int i0 = (blockIdx.x % tiles) * 64;
    const int b  = bh / Hc, h = bh % Hc;
    const int tid = threadIdx.x, wid = tid >> 5, lane = tid & 31;
    const int n16 = lane & 15, hi = lane >> 4;
    const int iw = i0 + wid * 16;

    const float invtemp = 1.0f / fmaxf(temp_p[0], 0.5f);

    __shared__ _Float16 kmS[32][KMLD];      // [key][d] km chunk (TDM target)
    __shared__ _Float16 Vt[64][34];         // [dh][key] transposed V chunk
    __shared__ _Float16 Ps[4][16][34];      // per-wave P scratch [m][k]

    v16h qaF = load_fragA(qa_h + ((size_t)bh * Tt + iw) * Dc, Dc, lane);
    v16h UF[Rc];
#pragma unroll
    for (int r = 0; r < Rc; ++r)
        UF[r] = load_fragA(U_h + (((size_t)bh * Rc + r) * Tt + iw) * Dc, Dc, lane);

    float rb[8];
#pragma unroll
    for (int e = 0; e < 8; ++e) rb[e] = rowbias[bh * Tt + iw + e + 8 * hi];

    v8f O[4];
    float m_i[8], l_i[8];
#pragma unroll
    for (int j = 0; j < 4; ++j)
#pragma unroll
        for (int e = 0; e < 8; ++e) O[j][e] = 0.0f;
#pragma unroll
    for (int e = 0; e < 8; ++e) { m_i[e] = -1e30f; l_i[e] = 0.0f; }

    const int nchunks = i0 / 32 + 2;        // keys 0 .. i0+63
    for (int c = 0; c < nchunks; ++c) {
        const int j0 = c * 32;

        __syncthreads();                    // previous kmS/Vt consumers done

#if HAVE_TDM
        if (wid == 0) {
            unsigned long long ga = (unsigned long long)(uintptr_t)km_h
                                  + (((unsigned long long)bh * Tt + (unsigned)j0) * Dc) * 2ull;
            unsigned int ldsb = (unsigned int)(unsigned long long)(uintptr_t)&kmS[0][0];
            u32x4 g0;
            g0[0] = 1u;                                   // count=1
            g0[1] = ldsb;                                 // lds_addr
            g0[2] = (unsigned int)ga;                     // global_addr[31:0]
            g0[3] = (unsigned int)((ga >> 32) & 0x1FFFFFFull) | (2u << 30); // addr hi | type=2
            i32x8 g1;
            g1[0] = (int)((1u << 16) | (1u << 20) | (3u << 22) | (1u << 25));
            g1[1] = (int)(32u << 16);                     // tensor_dim0 = 32
            g1[2] = (int)((unsigned)Tt << 16);            // tensor_dim1 = 512
            g1[3] = (int)(32u << 16);                     // tile_dim0 = 32
            g1[4] = (int)32u;                             // tile_dim1 = 32
            g1[5] = (int)32u;                             // tensor_dim0_stride = 32
            g1[6] = 0;
            g1[7] = 0;
            i32x4 gz; gz[0] = 0; gz[1] = 0; gz[2] = 0; gz[3] = 0;
#if __clang_major__ >= 23
            i32x8 gx;
#pragma unroll
            for (int e = 0; e < 8; ++e) gx[e] = 0;
            __builtin_amdgcn_tensor_load_to_lds(g0, g1, gz, gz, gx, 0);
#else
            __builtin_amdgcn_tensor_load_to_lds(g0, g1, gz, gz, 0);
#endif
        }
#else
#pragma unroll
        for (int it = 0; it < 2; ++it) {
            int i4 = tid + it * 128;
            int row = i4 >> 3, d4 = (i4 & 7) * 4;
            *(v4h*)&kmS[row][d4] =
                *(const v4h*)(km_h + ((size_t)bh * Tt + j0 + row) * Dc + d4);
        }
#endif
        // stage V chunk (f32 -> f16, transposed) while TDM runs
#pragma unroll
        for (int it = 0; it < 4; ++it) {
            int i4 = tid + it * 128;
            int key = i4 >> 4;
            int dh4 = (i4 & 15) * 4;
            v4f v4 = *(const v4f*)(Vbuf + (size_t)(b * Tt + j0 + key) * DMc + h * DHc + dh4);
#pragma unroll
            for (int e = 0; e < 4; ++e) Vt[dh4 + e][key] = (_Float16)v4[e];
        }
#if HAVE_TDM
        if (wid == 0) __builtin_amdgcn_s_wait_tensorcnt((short)0);
#endif
        __syncthreads();

        v8f   S[2];
        float pr[2][8];
#pragma unroll
        for (int cn = 0; cn < 2; ++cn) {
            const int jb = j0 + cn * 16;
            v16h Bkm = load_fragB(&kmS[cn * 16][0], KMLD, lane);

            v8f sqa;
#pragma unroll
            for (int e = 0; e < 8; ++e) sqa[e] = 0.0f;
            sqa = wmma_f16(qaF, Bkm, sqa);

            const float k2n = k2g[bh * Tt + jb + n16];
            float dist[8];
#pragma unroll
            for (int e = 0; e < 8; ++e) dist[e] = rb[e] + k2n - 2.0f * sqa[e];

#pragma unroll
            for (int r = 0; r < Rc; ++r) {
                v8f uk;
#pragma unroll
                for (int e = 0; e < 8; ++e) uk[e] = 0.0f;
                uk = wmma_f16(UF[r], Bkm, uk);
#pragma unroll
                for (int e = 0; e < 8; ++e) dist[e] += uk[e] * uk[e];
            }
#pragma unroll
            for (int e = 0; e < 8; ++e) {
                float s = -fmaxf(dist[e], 0.0f) * invtemp;
                int gi = iw + e + 8 * hi;
                int gj = jb + n16;
                S[cn][e] = (gj <= gi) ? s : -1e30f;
            }
        }

#pragma unroll
        for (int e = 0; e < 8; ++e) {
            float tm = redmax16(fmaxf(S[0][e], S[1][e]));
            float mn = fmaxf(m_i[e], tm);
            float sc = __expf(m_i[e] - mn);
            float p0 = __expf(S[0][e] - mn);
            float p1 = __expf(S[1][e] - mn);
            float rs = redsum16(p0 + p1);
            l_i[e] = l_i[e] * sc + rs;
            m_i[e] = mn;
#pragma unroll
            for (int j = 0; j < 4; ++j) O[j][e] *= sc;
            pr[0][e] = p0;
            pr[1][e] = p1;
        }

#pragma unroll
        for (int cn = 0; cn < 2; ++cn)
#pragma unroll
            for (int e = 0; e < 8; ++e)
                Ps[wid][e + 8 * hi][cn * 16 + n16] = (_Float16)pr[cn][e];

        v16h Pfrag = load_fragA(&Ps[wid][0][0], 34, lane);
#pragma unroll
        for (int j = 0; j < 4; ++j) {
            v16h Bv = load_fragB(&Vt[j * 16][0], 34, lane);
            O[j] = wmma_f16(Pfrag, Bv, O[j]);
        }
    }

#pragma unroll
    for (int e = 0; e < 8; ++e) {
        float inv = 1.0f / l_i[e];
        int row = iw + e + 8 * hi;
#pragma unroll
        for (int j = 0; j < 4; ++j)
            attn_out[(size_t)(b * Tt + row) * DMc + h * DHc + j * 16 + n16] = O[j][e] * inv;
    }
}

// ===================================================================
extern "C" void kernel_launch(void* const* d_in, const int* in_sizes, int n_in,
                              void* d_out, int out_size, void* d_ws, size_t ws_size,
                              hipStream_t stream) {
    (void)in_sizes; (void)n_in; (void)out_size; (void)ws_size;
    const float* x    = (const float*)d_in[0];
    const float* Wq   = (const float*)d_in[1];
    const float* Wk   = (const float*)d_in[2];
    const float* Wv   = (const float*)d_in[3];
    const float* Wo   = (const float*)d_in[4];
    const float* Wqm  = (const float*)d_in[5];
    const float* Wkm  = (const float*)d_in[6];
    const float* Wmet = (const float*)d_in[7];
    const float* temp = (const float*)d_in[8];
    float* out = (float*)d_out;

    char* ws = (char*)d_ws;
    float*     Qb    = (float*)(ws);
    float*     Kb    = (float*)(ws + ((size_t)4  << 20));
    float*     Vb    = (float*)(ws + ((size_t)8  << 20));
    float*     AOb   = (float*)(ws + ((size_t)12 << 20));
    _Float16*  km_h  = (_Float16*)(ws + ((size_t)16 << 20));
    _Float16*  qa_h  = (_Float16*)(ws + ((size_t)17 << 20));
    _Float16*  U_h   = (_Float16*)(ws + ((size_t)18 << 20));
    float*     rowb  = (float*)(ws + ((size_t)22 << 20));
    float*     k2g   = (float*)(ws + ((size_t)22 << 20) + (1 << 17));

    const int M = Bc * Tt, N = DMc, K = DMc;

    drm_gemm_f16w<<<dim3(64), dim3(256), 0, stream>>>(x, Wq, Qb, M, N, K);
    drm_gemm_f16w<<<dim3(64), dim3(256), 0, stream>>>(x, Wk, Kb, M, N, K);
    drm_gemm_f16w<<<dim3(64), dim3(256), 0, stream>>>(x, Wv, Vb, M, N, K);

    drm_metric<<<dim3((Bc * Hc * Tt) / 256), dim3(256), 0, stream>>>(
        Qb, Kb, Wqm, Wkm, Wmet, qa_h, km_h, U_h, rowb, k2g);

    drm_attn<<<dim3(Bc * Hc * (Tt / 64)), dim3(128), 0, stream>>>(
        qa_h, km_h, U_h, rowb, k2g, Vb, temp, AOb);

    drm_gemm_f16w<<<dim3(64), dim3(256), 0, stream>>>(AOb, Wo, out, M, N, K);
}
